// O2RNNlayer_21861383536865
// MI455X (gfx1250) — compile-verified, matrix operand order checked by reading, not run on hardware
//
#include <hip/hip_runtime.h>
#include <stdint.h>

#define NT 512      // timesteps
#define NB 64       // batch
#define NK 16       // in features
#define NO 256      // out features
#define NWG 16      // workgroups, each owns 16 output columns
#define THREADS 128 // 4 waves of 32

typedef __attribute__((ext_vector_type(16))) __bf16 v16bf;
typedef __attribute__((ext_vector_type(8)))  float  v8f;

union V16u { uint32_t u[8]; v16bf v; };

__device__ inline uint32_t pk_mul_bf16(uint32_t a, uint32_t b) {
  uint32_t d;
  asm volatile("v_pk_mul_bf16 %0, %1, %2" : "=v"(d) : "v"(a), "v"(b));
  return d;
}

// Per-launch init: zero the 512 barrier counters, seed h_buf[0] = bf16(hidden).
__global__ void o2rnn_init(const float* __restrict__ hidden,
                           uint32_t* __restrict__ cnt,
                           __bf16* __restrict__ hbuf0) {
  int i = blockIdx.x * blockDim.x + threadIdx.x;
  if (i < NT) cnt[i] = 0u;
  if (i < NB * NO) hbuf0[i] = (__bf16)hidden[i];
}

__global__ __launch_bounds__(THREADS, 1)
void o2rnn_main(const float* __restrict__ x,       // (512,64,16) f32
                const float* __restrict__ weight,  // (16, 65536) f32
                float* __restrict__ out,           // h_last (64*256) + output (512*64*256)
                __bf16* __restrict__ hbuf,         // 2 * 64*256 bf16 double buffer (d_ws)
                uint32_t* __restrict__ cnt) {      // 512 barrier counters (d_ws)
  __shared__ __bf16 wlds[NK * 16 * NO]; // 128 KB: this WG's weight slice, bf16
  __shared__ __bf16 hbf[NB * NO];       // 32 KB: current h, bf16

  const int wg   = blockIdx.x;   // output-column chunk: i in [wg*16, wg*16+16)
  const int tid  = threadIdx.x;
  const int lane = tid & 31;
  const int wv   = tid >> 5;     // wave id = mtile (batch rows wv*16 .. wv*16+15)
  const int il   = lane & 15;
  const int hi   = lane >> 4;

  // One-time: stage + convert weight slice. wlds[(k*16+i2)*256 + j] = weight[k][ (wg*16+i2)*256 + j ]
  for (int rr = tid; rr < NK * 16; rr += THREADS) {
    const int k  = rr >> 4;
    const int i2 = rr & 15;
    const float4* src = (const float4*)(weight + (size_t)k * (NO * NO) + (size_t)(wg * 16 + i2) * NO);
    __bf16* dst = &wlds[rr * NO];
    #pragma unroll 4
    for (int j4 = 0; j4 < NO / 4; ++j4) {
      float4 f = src[j4];
      dst[4 * j4 + 0] = (__bf16)f.x;
      dst[4 * j4 + 1] = (__bf16)f.y;
      dst[4 * j4 + 2] = (__bf16)f.z;
      dst[4 * j4 + 3] = (__bf16)f.w;
    }
  }

  const int bA = wv * 16 + il;   // A-matrix row this lane supplies (A layout: lane L -> M = L&15)
  const uint32_t* arow = (const uint32_t*)&hbf[bA * NO];

  for (int t = 0; t < NT; ++t) {
    // Stage current h (bf16) from global double buffer into LDS.
    const __bf16* hsrc = hbuf + (size_t)(t & 1) * (NB * NO);
    for (int e = tid * 8; e < NB * NO; e += THREADS * 8)
      *(uint4*)&hbf[e] = *(const uint4*)&hsrc[e];
    __syncthreads();

    // GEMM: D[b, i] = sum over K=4096 of (x[t,b,k]*h[b,j]) * W[k, i*256+j]
    v8f acc = {};
    uint32_t xp = 0;
    #pragma unroll 4
    for (int c = 0; c < 128; ++c) {      // k-iter: K rows [c*32, c*32+32), k = c>>3, j = (c&7)*32 + kk
      const int k  = c >> 3;
      const int j0 = (c & 7) << 5;
      if ((c & 7) == 0) {                // x scalar changes only when k changes
        float xf = x[((size_t)t * NB + bA) * NK + k];
        __bf16 xb = (__bf16)xf;
        uint16_t us; __builtin_memcpy(&us, &xb, 2);
        xp = (uint32_t)us * 0x10001u;    // bf16 pair {x, x}
      }
      const int dw = (j0 >> 1) + hi * 4; // dword offset: elements j0 + hi*8 (and +16)
      const uint32_t* brow = (const uint32_t*)&wlds[(k * 16 + il) * NO];
      uint4 b0 = *(const uint4*)(brow + dw);
      uint4 b1 = *(const uint4*)(brow + dw + 8);
      uint4 a0 = *(const uint4*)(arow + dw);
      uint4 a1 = *(const uint4*)(arow + dw + 8);
      V16u A, B;
      A.u[0] = pk_mul_bf16(a0.x, xp); A.u[1] = pk_mul_bf16(a0.y, xp);
      A.u[2] = pk_mul_bf16(a0.z, xp); A.u[3] = pk_mul_bf16(a0.w, xp);
      A.u[4] = pk_mul_bf16(a1.x, xp); A.u[5] = pk_mul_bf16(a1.y, xp);
      A.u[6] = pk_mul_bf16(a1.z, xp); A.u[7] = pk_mul_bf16(a1.w, xp);
      B.u[0] = b0.x; B.u[1] = b0.y; B.u[2] = b0.z; B.u[3] = b0.w;
      B.u[4] = b1.x; B.u[5] = b1.y; B.u[6] = b1.z; B.u[7] = b1.w;
      acc = __builtin_amdgcn_wmma_f32_16x16x32_bf16(false, A.v, false, B.v,
                                                    (short)0, acc, false, false);
    }

    // tanh + publish. C/D layout: VGPR r, lane L -> M = r + 8*(L>=16), N = L&15.
    float*  outp  = out + (size_t)(NB * NO) + (size_t)t * (NB * NO);
    __bf16* hnext = hbuf + (size_t)((t + 1) & 1) * (NB * NO);
    #pragma unroll
    for (int r = 0; r < 8; ++r) {
      const int b   = wv * 16 + r + hi * 8;
      const int idx = b * NO + wg * 16 + il;
      float hv = tanhf(acc[r]);
      outp[idx]  = hv;
      hnext[idx] = (__bf16)hv;
      if (t == NT - 1) out[idx] = hv;   // h_last lives at the front of d_out
    }

    // One device-wide barrier per step (per-step counter, re-zeroed each launch).
    __threadfence();
    __syncthreads();
    if (tid == 0) {
      __hip_atomic_fetch_add(&cnt[t], 1u, __ATOMIC_RELEASE, __HIP_MEMORY_SCOPE_AGENT);
      while (__hip_atomic_load(&cnt[t], __ATOMIC_ACQUIRE, __HIP_MEMORY_SCOPE_AGENT) < NWG)
        __builtin_amdgcn_s_sleep(2);
    }
    __syncthreads();
  }
}

extern "C" void kernel_launch(void* const* d_in, const int* in_sizes, int n_in,
                              void* d_out, int out_size, void* d_ws, size_t ws_size,
                              hipStream_t stream) {
  (void)in_sizes; (void)n_in; (void)out_size; (void)ws_size;
  const float* x      = (const float*)d_in[0];
  const float* hidden = (const float*)d_in[1];
  const float* weight = (const float*)d_in[2];
  float*   out = (float*)d_out;
  uint8_t* ws  = (uint8_t*)d_ws;
  uint32_t* cnt  = (uint32_t*)ws;             // 512 * 4 B
  __bf16*   hbuf = (__bf16*)(ws + 4096);      // 2 * 32 KB double buffer

  o2rnn_init<<<dim3((NB * NO) / 256), dim3(256), 0, stream>>>(hidden, cnt, hbuf);
  o2rnn_main<<<dim3(NWG), dim3(THREADS), 0, stream>>>(x, weight, out, hbuf, cnt);
}
